// DFFN_2765958939259
// MI455X (gfx1250) — matrix operand order repaired and checked by reading, not compiled, not run on hardware
//
#include <hip/hip_runtime.h>

typedef __attribute__((ext_vector_type(16))) _Float16 v16h;
typedef __attribute__((ext_vector_type(8)))  _Float16 v8h;
typedef __attribute__((ext_vector_type(8)))  float    v8f;
typedef __attribute__((ext_vector_type(4)))  float    v4f;

#define DIM 64
#define C2  340
#define HID 170
#define NB  4
#define HW  256

// ---------- WMMA fragment helpers (CDNA5 16x16x32 f16, wave32) ----------
// A-matrix 16x32 (MxK), row-major source with leading dim ldk:
//   lane: M = lane&15, Kbase = (lane>>4)*8; chunks K..K+7 and K+16..K+23
static __device__ __forceinline__ v16h ld_afrag(const _Float16* A, int ldk, int lane) {
  const int m  = lane & 15;
  const int kb = (lane >> 4) << 3;
  const _Float16* p = A + m * ldk + kb;
  v8h lo = *(const v8h*)(p);
  v8h hi = *(const v8h*)(p + 16);
  v16h r;
#pragma unroll
  for (int i = 0; i < 8; ++i) { r[i] = lo[i]; r[i + 8] = hi[i]; }
  return r;
}

// B-matrix 32x16 (KxN) sourced from a transposed (N-major) buffer BT[n][k]:
//   lane: N = lane&15, Kbase = (lane>>4)*8; chunks K..K+7 and K+16..K+23
static __device__ __forceinline__ v16h ld_btfrag(const _Float16* BT, int ldk, int lane) {
  const int n  = lane & 15;
  const int kb = (lane >> 4) << 3;
  const _Float16* p = BT + n * ldk + kb;
  v8h lo = *(const v8h*)(p);
  v8h hi = *(const v8h*)(p + 16);
  v16h r;
#pragma unroll
  for (int i = 0; i < 8; ++i) { r[i] = lo[i]; r[i + 8] = hi[i]; }
  return r;
}

static __device__ __forceinline__ v8f wmma_f16(v16h a, v16h b, v8f c) {
  return __builtin_amdgcn_wmma_f32_16x16x32_f16(false, a, false, b, (short)0, c, false, false);
}

// ---------- Kernel 0: build per-channel 64x64 circulant matrices from fft_filter ----------
// Effective spectral multiplier (matching numpy irfft2 semantics incl. non-Hermitian
// DC/Nyquist columns): Fsym = Herm2d(Fhat), g = real(ifft2(Fsym)), M_c[p][q] = g[(p-q) mod 8x8]
__global__ void dffn_prep_M(const float* __restrict__ fftf, _Float16* __restrict__ Mmat) {
  const int c = blockIdx.x;   // 0..339
  const int t = threadIdx.x;  // 0..63
  __shared__ float Fh[64];
  __shared__ float g[64];

  const int a = t >> 3, b = t & 7;
  float v;
  if (b <= 4) v = fftf[c * 40 + a * 5 + b];
  else        v = fftf[c * 40 + (((8 - a) & 7) * 5) + (8 - b)];
  Fh[t] = v;
  __syncthreads();

  const int u = t >> 3, vv = t & 7;
  const float ctab[8] = {1.f, 0.70710678f, 0.f, -0.70710678f,
                         -1.f, -0.70710678f, 0.f, 0.70710678f};
  float s = 0.f;
  for (int aa = 0; aa < 8; ++aa)
    for (int bb = 0; bb < 8; ++bb) {
      float fs = 0.5f * (Fh[aa * 8 + bb] + Fh[((8 - aa) & 7) * 8 + ((8 - bb) & 7)]);
      s += fs * ctab[(aa * u + bb * vv) & 7];
    }
  g[t] = s * (1.f / 64.f);
  __syncthreads();

  const int up = t >> 3, vp = t & 7;
  _Float16* Mrow = Mmat + (size_t)c * 4096 + (size_t)t * 64;
  for (int q = 0; q < 64; ++q) {
    int uq = q >> 3, vq = q & 7;
    Mrow[q] = (_Float16)g[((up - uq) & 7) * 8 + ((vp - vq) & 7)];
  }
}

// ---------- Kernel 1: fused project_in (WMMA) + per-patch spectral conv (WMMA) ----------
// WG: 256 thr (8 waves), 16 patches = 8 rows x 128 cols. Grid = 4b * 32pr * 2pcb = 256.
// LDS ~270KB -> 1 WG/WGP (2 waves/SIMD); launch_bounds(256,2) lifts the VGPR cap
// so the 8 cached A-fragments + 4 accumulators don't spill to scratch.
__global__ void __launch_bounds__(256, 2)
dffn_stage1(const float* __restrict__ x, const float* __restrict__ W_in,
            const _Float16* __restrict__ Mmat, _Float16* __restrict__ Y) {
  extern __shared__ __align__(16) char smem[];
  _Float16* Tlds = (_Float16*)smem;        // [c_local=64][patch=16][pix=64] (at LDS 0:
  _Float16* Xlds = Tlds + 64 * 1024;       //   scatter-store offsets fit DS imm field)
  _Float16* Alds = Xlds + 1024 * 64;       // [64][64] W_in slice (f16)

  const int t = threadIdx.x;
  const int lane = t & 31, wave = t >> 5;
  const int bi = blockIdx.x;
  const int b   = bi >> 6;
  const int pr  = (bi & 63) >> 1;  // patch row 0..31
  const int pcb = bi & 1;          // 0..1 (128-col block)
  const int h0 = pr * 8;
  const int w0 = pcb * 128;

  // Load X tile: 64 ch x (8 rows x 128 cols), f16, n = pix*16 + patch ordering
  {
    const int u     = t >> 5;         // row in patch 0..7
    const int cw    = (t & 31) * 4;   // col 0..124, step 4 (patch-constant per 4)
    const int patch = cw >> 3;
    const int vbase = cw & 7;
    for (int k = 0; k < 64; ++k) {
      const float* gp = x + (((size_t)(b * DIM + k) * HW + (h0 + u)) * HW + (w0 + cw));
      v4f xv = *(const v4f*)gp;
#pragma unroll
      for (int e = 0; e < 4; ++e) {
        int pix = u * 8 + vbase + e;
        Xlds[(pix * 16 + patch) * 64 + k] = (_Float16)xv[e];
      }
    }
  }
  __syncthreads();

  for (int cs = 0; cs < 6; ++cs) {   // 6 slices of 64 channels (340 padded to 384)
    for (int i = t; i < 64 * 64; i += 256) {
      int r = i >> 6, k = i & 63;
      int c = cs * 64 + r;
      Alds[i] = (c < C2) ? (_Float16)W_in[c * 64 + k] : (_Float16)0.f;
    }
    __syncthreads();

    // GEMM1: T[64][1024] = Wslice[64][64] @ X[64][1024]
    v16h a0[4], a1[4];
#pragma unroll
    for (int mt = 0; mt < 4; ++mt) {
      a0[mt] = ld_afrag(Alds + mt * 16 * 64 + 0,  64, lane);
      a1[mt] = ld_afrag(Alds + mt * 16 * 64 + 32, 64, lane);
    }
    for (int nt = wave; nt < 64; nt += 8) {
      v16h b0 = ld_btfrag(Xlds + nt * 16 * 64 + 0,  64, lane);
      v16h b1 = ld_btfrag(Xlds + nt * 16 * 64 + 32, 64, lane);
      const int n = nt * 16 + (lane & 15);
      const int pix = n >> 4, patch = n & 15;
      const int cb  = ((lane >> 4) << 3);
      // issue all 8 WMMAs back-to-back (independent accumulators: no D->A/B hazard),
      // then drain the stores; store VALU hides the WMMA->VALU co-exec window
      v8f acc[4];
#pragma unroll
      for (int mt = 0; mt < 4; ++mt) {
        v8f z = {};
        z = wmma_f16(a0[mt], b0, z);
        acc[mt] = wmma_f16(a1[mt], b1, z);
      }
#pragma unroll
      for (int mt = 0; mt < 4; ++mt) {
        _Float16* tp = Tlds + (size_t)(mt * 16 + cb) * 1024 + patch * 64 + pix;
#pragma unroll
        for (int r = 0; r < 8; ++r)
          tp[r * 1024] = (_Float16)acc[mt][r];   // ds_store_b16, offset = r*2048B (imm)
      }
    }
    __syncthreads();

    // GEMM2: per channel, Yc[64pix][16patch] = M_c[64][64] @ Tc[64][16]
    for (int cl = wave; cl < 64; cl += 8) {
      int c = cs * 64 + cl;
      if (c >= C2) continue;                      // wave-uniform
      const _Float16* Mc = Mmat + (size_t)c * 4096;
      __builtin_prefetch(Mc + 4096, 0, 0);        // global_prefetch for next channel
      const _Float16* Tc = Tlds + cl * 1024;
      v16h tb0 = ld_btfrag(Tc + 0,  64, lane);
      v16h tb1 = ld_btfrag(Tc + 32, 64, lane);
      const int patch = lane & 15;
      const int hi    = lane >> 4;
      v16h ma0[4], ma1[4];
#pragma unroll
      for (int mt = 0; mt < 4; ++mt) {
        ma0[mt] = ld_afrag(Mc + mt * 16 * 64 + 0,  64, lane);
        ma1[mt] = ld_afrag(Mc + mt * 16 * 64 + 32, 64, lane);
      }
      v8f acc[4];
#pragma unroll
      for (int mt = 0; mt < 4; ++mt) {
        v8f z = {};
        z = wmma_f16(ma0[mt], tb0, z);
        acc[mt] = wmma_f16(ma1[mt], tb1, z);
      }
#pragma unroll
      for (int mt = 0; mt < 4; ++mt) {
        v8h pk;
#pragma unroll
        for (int r = 0; r < 8; ++r) pk[r] = (_Float16)acc[mt][r];
        const int uloc = mt * 2 + hi;   // pix_out>>3; r = pix_out&7 (contiguous cols)
        _Float16* yp = Y + (((size_t)(b * C2 + c) * HW + (h0 + uloc)) * HW + (w0 + patch * 8));
        *(v8h*)yp = pk;                 // 16B packed store
      }
    }
    __syncthreads();
  }
}

// ---------- Kernel 2: depthwise 3x3 + exact GELU gate + project_out (WMMA) ----------
// WG: 256 thr (8 waves), tile 8 rows x 32 cols. Grid = 4b * 32 * 8 = 1024.
// Halo tiles (10x40 per channel, zero-padded borders) are DMA'd into LDS with
// CDNA5 async global->LDS loads, double-buffered across channel iterations and
// drained with s_wait_asynccnt.
__global__ void dffn_stage2(const _Float16* __restrict__ Y, const float* __restrict__ W_dw,
                            const float* __restrict__ W_out, float* __restrict__ out) {
  extern __shared__ __align__(16) char smem[];
  _Float16* Hlds = (_Float16*)smem;             // [n=256][k=192] gated hidden (B^T for GEMM3)
  _Float16* Alds = Hlds + 256 * 192;            // [64][192] W_out padded f16
  _Float16* Ybuf = Alds + 64 * 192;             // [2 dbuf][2 chan][10 rows][40 cols] f16
  float*    wdw  = (float*)(Ybuf + 1600);       // [340][9]

  const int t = threadIdx.x;
  const int lane = t & 31, wave = t >> 5;
  const int bi = blockIdx.x;
  const int b  = bi >> 8;
  const int hb = (bi & 255) >> 3;
  const int wb = bi & 7;
  const int rr = t >> 5, cc = t & 31;
  const size_t plane = (size_t)HW * HW;

  for (int i = t; i < 64 * 192; i += 256) {
    int o = i / 192, k = i % 192;
    Alds[i] = (k < HID) ? (_Float16)W_out[o * HID + k] : (_Float16)0.f;
  }
  for (int i = t; i < C2 * 9; i += 256) wdw[i] = W_dw[i];
  for (int k = HID; k < 192; ++k) Hlds[t * 192 + k] = (_Float16)0.f;   // K pad
  for (int i = t; i < 1600; i += 256) Ybuf[i] = (_Float16)0.f;         // halo zero-pad
  __syncthreads();

  const uint32_t ybase = (uint32_t)(uintptr_t)(void*)Ybuf;  // LDS byte offset (addr[31:0])

  // Issue the 10x40-f16 halos of channels (c1, c1+HID) into buffer dbuf.
  // 200 8-byte chunks -> 7 wave32 async-load instructions (EXEC-masked at borders;
  // masked chunks stay zero from the one-time init). Wave 0 only.
  auto issue_halo = [&](int c1, int dbuf) {
#pragma unroll
    for (int j = 0; j < 7; ++j) {
      int item = j * 32 + lane;
      int ch    = item / 100;
      int rem   = item - ch * 100;
      int row   = rem / 10;
      int chunk = rem - row * 10;
      int hh = hb * 8 - 1 + row;
      int ww = wb * 32 - 4 + chunk * 4;       // 4-f16 chunks, border-aligned
      bool ok = (item < 200) && ((unsigned)hh < 256u) && ((unsigned)ww <= 252u);
      if (ok) {
        const _Float16* gp =
            Y + (size_t)(b * C2 + c1 + ch * HID) * plane + (size_t)hh * HW + ww;
        uint32_t laddr =
            ybase + (uint32_t)(((((dbuf * 2 + ch) * 10) + row) * 40 + chunk * 4) * 2);
        asm volatile("global_load_async_to_lds_b64 %0, %1, off"
                     :: "v"(laddr), "v"(gp) : "memory");
      }
    }
  };

  if (wave == 0) issue_halo(0, 0);

  for (int c1 = 0; c1 < HID; ++c1) {
    const int cur = c1 & 1;
    if (wave == 0) {
      if (c1 + 1 < HID) {
        issue_halo(c1 + 1, cur ^ 1);
        // 7 newest in flight belong to the next iteration; current buffer is done
        asm volatile("s_wait_asynccnt 0x7" ::: "memory");
      } else {
        asm volatile("s_wait_asynccnt 0x0" ::: "memory");
      }
    }
    __syncthreads();  // current halo visible to all waves

    const _Float16* Y1l = Ybuf + (cur * 2 + 0) * 400;
    const _Float16* Y2l = Ybuf + (cur * 2 + 1) * 400;
    const float* w1 = wdw + c1 * 9;
    const float* w2 = wdw + (c1 + HID) * 9;
    float z1 = 0.f, z2 = 0.f;
#pragma unroll
    for (int di = 0; di < 3; ++di) {
#pragma unroll
      for (int dj = 0; dj < 3; ++dj) {
        int off = (rr + di) * 40 + (cc + 3 + dj);
        float wt1 = w1[di * 3 + dj], wt2 = w2[di * 3 + dj];
        z1 += wt1 * (float)Y1l[off];
        z2 += wt2 * (float)Y2l[off];
      }
    }
    float ge = 0.5f * z1 * (1.f + erff(z1 * 0.70710678118f));  // exact erf GELU
    Hlds[t * 192 + c1] = (_Float16)(ge * z2);
    __syncthreads();  // compute done -> next prefetch may overwrite buffer `cur`
  }

  // GEMM3: Out[64][256px] = Wout[64][192] @ H[192][256]
  for (int nt = wave * 2; nt < wave * 2 + 2; ++nt) {
    v8f acc[4];
#pragma unroll
    for (int mt = 0; mt < 4; ++mt) { v8f z = {}; acc[mt] = z; }
    for (int ks = 0; ks < 6; ++ks) {
      v16h bf = ld_btfrag(Hlds + nt * 16 * 192 + ks * 32, 192, lane);
#pragma unroll
      for (int mt = 0; mt < 4; ++mt) {
        v16h af = ld_afrag(Alds + mt * 16 * 192 + ks * 32, 192, lane);
        acc[mt] = wmma_f16(af, bf, acc[mt]);
      }
    }
    const int n   = nt * 16 + (lane & 15);
    const int orr = n >> 5, occ = n & 31;
    const int hi  = lane >> 4;
#pragma unroll
    for (int mt = 0; mt < 4; ++mt) {
#pragma unroll
      for (int r = 0; r < 8; ++r) {
        int o = mt * 16 + 8 * hi + r;
        out[(((size_t)(b * DIM + o)) * HW + (hb * 8 + orr)) * HW + (wb * 32 + occ)] = acc[mt][r];
      }
    }
  }
}

extern "C" void kernel_launch(void* const* d_in, const int* in_sizes, int n_in,
                              void* d_out, int out_size, void* d_ws, size_t ws_size,
                              hipStream_t stream) {
  (void)in_sizes; (void)n_in; (void)out_size; (void)ws_size;
  const float* x     = (const float*)d_in[0];
  const float* W_in  = (const float*)d_in[1];
  const float* W_dw  = (const float*)d_in[2];
  const float* fftf  = (const float*)d_in[3];
  const float* W_out = (const float*)d_in[4];
  float* out = (float*)d_out;

  char* ws = (char*)d_ws;
  _Float16* Y    = (_Float16*)ws;                                   // 4*340*256*256 f16 (~170MB)
  _Float16* Mmat = (_Float16*)(ws + (size_t)NB * C2 * HW * HW * 2); // 340*64*64 f16 (~2.8MB)

  dffn_prep_M<<<C2, 64, 0, stream>>>(fftf, Mmat);

  size_t lds1 = (size_t)(1024 * 64 + 64 * 1024 + 64 * 64) * sizeof(_Float16); // ~270KB
  dffn_stage1<<<NB * 32 * 2, 256, lds1, stream>>>(x, W_in, Mmat, Y);

  size_t lds2 = (size_t)(256 * 192 + 64 * 192 + 1600) * sizeof(_Float16)
              + (size_t)C2 * 9 * sizeof(float);                               // ~138KB
  dffn_stage2<<<NB * 32 * 8, 256, lds2, stream>>>(Y, W_dw, W_out, out);
}
